// MultiHeadSelfAttention_86517821215074
// MI455X (gfx1250) — matrix operand order repaired and checked by reading, not compiled, not run on hardware
//
#include <hip/hip_runtime.h>

// MI455X / gfx1250 multi-head self-attention, bf16 WMMA + f32 online softmax.
// B=2, T=2048, C=1024, H=16, D=64.
// v3: software-pipelined GEMM inner loops; attention double-buffers K/V tiles
//     in LDS via GLOBAL_LOAD_ASYNC_TO_LDS_B128 (ASYNCcnt) overlapped with WMMA.

typedef __bf16 bf16;
typedef __attribute__((ext_vector_type(16))) __bf16 bf16x16;
typedef __attribute__((ext_vector_type(8)))  __bf16 bf16x8;
typedef __attribute__((ext_vector_type(8)))  float  f32x8;

constexpr int B = 2, T = 2048, C = 1024, H = 16, D = 64;
constexpr int BT = B * T;

__device__ __forceinline__ f32x8 wmma_bf16(bf16x16 a, bf16x16 b, f32x8 c) {
  // D = A(16x32) * B(32x16) + C, f32 accumulate
  return __builtin_amdgcn_wmma_f32_16x16x32_bf16(
      /*neg_a=*/false, a, /*neg_b=*/false, b,
      /*c_mod=*/(short)0, c, /*reuse_a=*/false, /*reuse_b=*/false);
}

// A-matrix fragment (16x32 bf16, row M = lane&15). Per ISA 7.12.2 a lane needs
// K = base..base+7 and base+16..base+23 with base = (lane>=16)*8 -> two 16B loads.
__device__ __forceinline__ bf16x16 load_a_frag(const bf16* row, int lane) {
  const int klo = (lane >= 16) ? 8 : 0;
  bf16x8 lo = *(const bf16x8*)(row + klo);
  bf16x8 hi = *(const bf16x8*)(row + 16 + klo);
  bf16x16 a;
#pragma unroll
  for (int i = 0; i < 8; ++i) { a[i] = lo[i]; a[i + 8] = hi[i]; }
  return a;
}

// B-matrix fragment (32x16 bf16, col N = lane&15). A lane needs K = base..base+7
// and base+16..base+23 with base = (lane>=16)*8, from a [N][K] (K-contiguous) layout.
__device__ __forceinline__ bf16x16 load_b_frag(const bf16* colK, int lane) {
  const int koff = (lane >= 16) ? 8 : 0;
  bf16x8 lo = *(const bf16x8*)(colK + koff);
  bf16x8 hi = *(const bf16x8*)(colK + 16 + koff);
  bf16x16 b;
#pragma unroll
  for (int i = 0; i < 8; ++i) { b[i] = lo[i]; b[i + 8] = hi[i]; }
  return b;
}

// ---------------------------------------------------------------------------
// Async global -> LDS staging (CDNA5 Tensor/Async path, ASYNCcnt-tracked).
// INST_OFFSET is added to BOTH the LDS and global address (ISA 08 §4.4), so a
// fixed (vdst, vaddr, saddr) moves a whole per-lane row with immediate offsets.
// ---------------------------------------------------------------------------
#define ASYNC_LDS_B128(dst, voff, sbase, IMM)                              \
  asm volatile("global_load_async_to_lds_b128 %0, %1, %2 offset:" #IMM     \
               :: "v"(dst), "v"(voff), "s"(sbase) : "memory")

__device__ __forceinline__ void stage_row_128B(const bf16* gbase, unsigned voff,
                                               unsigned dst) {
  const unsigned long long sb = (unsigned long long)(uintptr_t)gbase;
  ASYNC_LDS_B128(dst, voff, sb, 0);
  ASYNC_LDS_B128(dst, voff, sb, 16);
  ASYNC_LDS_B128(dst, voff, sb, 32);
  ASYNC_LDS_B128(dst, voff, sb, 48);
  ASYNC_LDS_B128(dst, voff, sb, 64);
  ASYNC_LDS_B128(dst, voff, sb, 80);
  ASYNC_LDS_B128(dst, voff, sb, 96);
  ASYNC_LDS_B128(dst, voff, sb, 112);
}

__device__ __forceinline__ void stage_row_64B(const bf16* gbase, unsigned voff,
                                              unsigned dst) {
  const unsigned long long sb = (unsigned long long)(uintptr_t)gbase;
  ASYNC_LDS_B128(dst, voff, sb, 0);
  ASYNC_LDS_B128(dst, voff, sb, 16);
  ASYNC_LDS_B128(dst, voff, sb, 32);
  ASYNC_LDS_B128(dst, voff, sb, 48);
}

__device__ __forceinline__ void wait_async0() {
  asm volatile("s_wait_asynccnt 0x0" ::: "memory");
}

// ---------------------------------------------------------------------------
// fp32 -> bf16 conversion (grid-stride)
// ---------------------------------------------------------------------------
__global__ void cvt_bf16(const float* __restrict__ s, bf16* __restrict__ d, int n) {
  for (int i = blockIdx.x * blockDim.x + threadIdx.x; i < n;
       i += gridDim.x * blockDim.x)
    d[i] = (bf16)s[i];
}

// ---------------------------------------------------------------------------
// Fused QKV projection: one wave computes a 16x64 tile of x @ W^T + b,
// 2-stage register pipeline so loads run one K-step ahead of the WMMAs.
// mat 0 -> Q [B,H,T,D]; mat 1 -> K*(1/sqrt(D)) [B,H,T,D]; mat 2 -> V^T [B,H,D,T]
// ---------------------------------------------------------------------------
__global__ void __launch_bounds__(32)
qkv_proj(const bf16* __restrict__ xb,
         const bf16* __restrict__ Wq, const bf16* __restrict__ Wk,
         const bf16* __restrict__ Wv,
         const float* __restrict__ bq, const float* __restrict__ bk,
         const float* __restrict__ bv,
         bf16* __restrict__ Qo, bf16* __restrict__ Ko, bf16* __restrict__ Vto) {
  const int lane = threadIdx.x & 31;
  const int n    = lane & 15;
  const int half = lane >> 4;
  const int m0   = blockIdx.x * 16;
  const int n0   = blockIdx.y * 64;
  const int mat  = blockIdx.z;

  const bf16* W     = (mat == 0) ? Wq : (mat == 1) ? Wk : Wv;
  const float* bias = (mat == 0) ? bq : (mat == 1) ? bk : bv;

  const bf16* arow = xb + (size_t)(m0 + n) * C;        // A row M = lane&15
  const bf16* b0   = W + (size_t)(n0 +  0 + n) * C;    // [N][K] layout rows
  const bf16* b1   = W + (size_t)(n0 + 16 + n) * C;
  const bf16* b2   = W + (size_t)(n0 + 32 + n) * C;
  const bf16* b3   = W + (size_t)(n0 + 48 + n) * C;

  f32x8 acc0 = {}, acc1 = {}, acc2 = {}, acc3 = {};
  bf16x16 a  = load_a_frag(arow, lane);
  bf16x16 f0 = load_b_frag(b0, lane);
  bf16x16 f1 = load_b_frag(b1, lane);
  bf16x16 f2 = load_b_frag(b2, lane);
  bf16x16 f3 = load_b_frag(b3, lane);
#pragma unroll 4
  for (int kk = 32; kk < C; kk += 32) {
    if ((kk & 255) == 0 && kk + 512 < C) {
      __builtin_prefetch(arow + kk + 512, 0, 3);
      __builtin_prefetch(b0 + kk + 512, 0, 3);
      __builtin_prefetch(b2 + kk + 512, 0, 3);
    }
    const bf16x16 an = load_a_frag(arow + kk, lane);
    const bf16x16 g0 = load_b_frag(b0 + kk, lane);
    const bf16x16 g1 = load_b_frag(b1 + kk, lane);
    const bf16x16 g2 = load_b_frag(b2 + kk, lane);
    const bf16x16 g3 = load_b_frag(b3 + kk, lane);
    acc0 = wmma_bf16(a, f0, acc0);
    acc1 = wmma_bf16(a, f1, acc1);
    acc2 = wmma_bf16(a, f2, acc2);
    acc3 = wmma_bf16(a, f3, acc3);
    a = an; f0 = g0; f1 = g1; f2 = g2; f3 = g3;
  }
  acc0 = wmma_bf16(a, f0, acc0);
  acc1 = wmma_bf16(a, f1, acc1);
  acc2 = wmma_bf16(a, f2, acc2);
  acc3 = wmma_bf16(a, f3, acc3);

  const int hh = n0 >> 6;          // n0 is 64-aligned: whole tile in one head
  const int bb = m0 / T;           // 16-row tile never straddles batch rows
  const int t0 = m0 % T;
  f32x8* accs[4] = {&acc0, &acc1, &acc2, &acc3};
#pragma unroll
  for (int nb = 0; nb < 4; ++nb) {
    const int   d    = nb * 16 + n;
    const float bv_n = bias[n0 + d];
#pragma unroll
    for (int r = 0; r < 8; ++r) {
      const int t = t0 + r + 8 * half;
      const float v = (*accs[nb])[r] + bv_n;
      if (mat == 0) {
        Qo[(((size_t)(bb * H + hh)) * T + t) * D + d] = (bf16)v;
      } else if (mat == 1) {
        Ko[(((size_t)(bb * H + hh)) * T + t) * D + d] = (bf16)(v * 0.125f);
      } else {
        Vto[(((size_t)(bb * H + hh)) * D + d) * T + t] = (bf16)v;
      }
    }
  }
}

// ---------------------------------------------------------------------------
// Online softmax update for one 16-row query tile against 32 keys.
// C/D layout: row = r + 8*half, cols n (s0) and n+16 (s1).
// ---------------------------------------------------------------------------
__device__ __forceinline__ void online_softmax(
    f32x8& s0, f32x8& s1, f32x8& mrow, f32x8& lrow,
    f32x8& o0, f32x8& o1, f32x8& o2, f32x8& o3,
    f32x8& p0, f32x8& p1, int qbase, int j0, int n, int half) {
#pragma unroll
  for (int r = 0; r < 8; ++r) {
    const int qi = qbase + r + 8 * half;
    if (j0 + n      > qi) s0[r] = -__builtin_inff();
    if (j0 + 16 + n > qi) s1[r] = -__builtin_inff();
  }
  f32x8 mb;
#pragma unroll
  for (int r = 0; r < 8; ++r) mb[r] = fmaxf(s0[r], s1[r]);
#pragma unroll
  for (int m = 1; m <= 8; m <<= 1)
#pragma unroll
    for (int r = 0; r < 8; ++r) mb[r] = fmaxf(mb[r], __shfl_xor(mb[r], m, 32));
  f32x8 rs;
#pragma unroll
  for (int r = 0; r < 8; ++r) {
    const float mn = fmaxf(mrow[r], mb[r]);
    const float al = __expf(mrow[r] - mn);
    p0[r] = __expf(s0[r] - mn);
    p1[r] = __expf(s1[r] - mn);
    lrow[r] *= al;
    o0[r] *= al; o1[r] *= al; o2[r] *= al; o3[r] *= al;
    mrow[r] = mn;
    rs[r] = p0[r] + p1[r];
  }
#pragma unroll
  for (int m = 1; m <= 8; m <<= 1)
#pragma unroll
    for (int r = 0; r < 8; ++r) rs[r] += __shfl_xor(rs[r], m, 32);
#pragma unroll
  for (int r = 0; r < 8; ++r) lrow[r] += rs[r];
}

// ---------------------------------------------------------------------------
// Flash attention: one wave owns 32 query rows (two 16-row tiles A/B) of one
// (b,h). K/V tiles for the NEXT 32-key block stream global->LDS with async
// b128 copies (ASYNCcnt) while the current block's 16 WMMAs + softmax run.
// ---------------------------------------------------------------------------
__global__ void __launch_bounds__(32)
attn_fa(const bf16* __restrict__ Q, const bf16* __restrict__ K,
        const bf16* __restrict__ Vt, bf16* __restrict__ AO) {
  __shared__ __align__(16) bf16 Kbuf[2][32 * 64];  // [key row][d], 4 KB each
  __shared__ __align__(16) bf16 Vbuf[2][64 * 32];  // [d row][t],   4 KB each
  __shared__ __align__(16) bf16 Pbuf[32 * 32];     // P C-layout -> A-layout

  const int lane = threadIdx.x & 31;
  const int n    = lane & 15;
  const int half = lane >> 4;
  const int q0   = blockIdx.x * 32;
  const int bh   = blockIdx.y;
  const int b    = bh >> 4;   // H == 16
  const int h    = bh & 15;

  const bf16* Qb = Q  + (size_t)bh * T * D;
  const bf16* Kb = K  + (size_t)bh * T * D;
  const bf16* Vb = Vt + (size_t)bh * D * T;

  const unsigned ldsK[2] = {(unsigned)(uintptr_t)&Kbuf[0][0],
                            (unsigned)(uintptr_t)&Kbuf[1][0]};
  const unsigned ldsV[2] = {(unsigned)(uintptr_t)&Vbuf[0][0],
                            (unsigned)(uintptr_t)&Vbuf[1][0]};

  // Q fragments for both 16-row tiles, full head dim (K-steps d=0..31, 32..63)
  const bf16* qrA = Qb + (size_t)(q0 + n) * D;
  const bf16* qrB = Qb + (size_t)(q0 + 16 + n) * D;
  const bf16x16 qa0A = load_a_frag(qrA, lane), qa1A = load_a_frag(qrA + 32, lane);
  const bf16x16 qa0B = load_a_frag(qrB, lane), qa1B = load_a_frag(qrB + 32, lane);

  f32x8 oA0 = {}, oA1 = {}, oA2 = {}, oA3 = {};
  f32x8 oB0 = {}, oB1 = {}, oB2 = {}, oB3 = {};
  f32x8 mA, lA, mB, lB;
#pragma unroll
  for (int r = 0; r < 8; ++r) {
    mA[r] = -__builtin_inff(); lA[r] = 0.f;
    mB[r] = -__builtin_inff(); lB[r] = 0.f;
  }

  // --- stage first K/V tile (32 keys): lane owns key row `lane` (128 B) and
  //     V^T rows d=lane, d=lane+32 (64 B each) ---
  stage_row_128B(Kb, (unsigned)(lane * (D * 2)), ldsK[0] + lane * 128);
  stage_row_64B(Vb, (unsigned)(lane * (T * 2)), ldsV[0] + lane * 64);
  stage_row_64B(Vb, (unsigned)((lane + 32) * (T * 2)), ldsV[0] + 2048 + lane * 64);

  const int jend = q0 + 31;
  for (int j0 = 0; j0 <= jend; j0 += 32) {
    const int cur = (j0 >> 5) & 1;
    wait_async0();                      // current tile resident in LDS
    if (j0 + 32 <= jend) {              // kick off next tile while we compute
      const int nxt = cur ^ 1;
      const bf16* kg = Kb + (size_t)(j0 + 32) * D;
      const bf16* vg = Vb + (j0 + 32);
      stage_row_128B(kg, (unsigned)(lane * (D * 2)), ldsK[nxt] + lane * 128);
      stage_row_64B(vg, (unsigned)(lane * (T * 2)), ldsV[nxt] + lane * 64);
      stage_row_64B(vg, (unsigned)((lane + 32) * (T * 2)),
                    ldsV[nxt] + 2048 + lane * 64);
    }
    // ---- K fragments from LDS (col N=n -> key row n / n+16, d contiguous) ----
    const bf16* kt = &Kbuf[cur][0];
    const bf16x16 kb0lo = load_b_frag(kt + n * 64, lane);
    const bf16x16 kb0hi = load_b_frag(kt + n * 64 + 32, lane);
    const bf16x16 kb1lo = load_b_frag(kt + (16 + n) * 64, lane);
    const bf16x16 kb1hi = load_b_frag(kt + (16 + n) * 64 + 32, lane);
    // ---- scores: 8 WMMAs reuse the 4 K fragments ----
    f32x8 sA0 = {}, sA1 = {}, sB0 = {}, sB1 = {};
    sA0 = wmma_bf16(qa0A, kb0lo, sA0); sA0 = wmma_bf16(qa1A, kb0hi, sA0);
    sA1 = wmma_bf16(qa0A, kb1lo, sA1); sA1 = wmma_bf16(qa1A, kb1hi, sA1);
    sB0 = wmma_bf16(qa0B, kb0lo, sB0); sB0 = wmma_bf16(qa1B, kb0hi, sB0);
    sB1 = wmma_bf16(qa0B, kb1lo, sB1); sB1 = wmma_bf16(qa1B, kb1hi, sB1);
    // ---- mask + online softmax per tile ----
    f32x8 pA0, pA1, pB0, pB1;
    online_softmax(sA0, sA1, mA, lA, oA0, oA1, oA2, oA3, pA0, pA1, q0, j0, n, half);
    online_softmax(sB0, sB1, mB, lB, oB0, oB1, oB2, oB3, pB0, pB1, q0 + 16, j0, n, half);
    // ---- P (bf16) to LDS row-major [32][32] ----
#pragma unroll
    for (int r = 0; r < 8; ++r) {
      const int ra = (r + 8 * half) * 32;
      Pbuf[ra + n]            = (bf16)pA0[r];
      Pbuf[ra + n + 16]       = (bf16)pA1[r];
      Pbuf[ra + 512 + n]      = (bf16)pB0[r];   // rows 16..31
      Pbuf[ra + 512 + n + 16] = (bf16)pB1[r];
    }
    __syncthreads();
    const bf16x16 paA = load_a_frag(&Pbuf[n * 32], lane);
    const bf16x16 paB = load_a_frag(&Pbuf[(16 + n) * 32], lane);
    __syncthreads();
    // ---- O += P V : 8 WMMAs reuse 4 V fragments from LDS ----
    const bf16* vt = &Vbuf[cur][0];
    const bf16x16 vb0 = load_b_frag(vt + ( 0 + n) * 32, lane);
    const bf16x16 vb1 = load_b_frag(vt + (16 + n) * 32, lane);
    const bf16x16 vb2 = load_b_frag(vt + (32 + n) * 32, lane);
    const bf16x16 vb3 = load_b_frag(vt + (48 + n) * 32, lane);
    oA0 = wmma_bf16(paA, vb0, oA0); oA1 = wmma_bf16(paA, vb1, oA1);
    oA2 = wmma_bf16(paA, vb2, oA2); oA3 = wmma_bf16(paA, vb3, oA3);
    oB0 = wmma_bf16(paB, vb0, oB0); oB1 = wmma_bf16(paB, vb1, oB1);
    oB2 = wmma_bf16(paB, vb2, oB2); oB3 = wmma_bf16(paB, vb3, oB3);
  }

  // ---- normalize + merge heads: AO[b, t, h*64 + d] bf16 row-major ----
  bf16* dst = AO + (size_t)b * T * C;
#pragma unroll
  for (int r = 0; r < 8; ++r) {
    const int tA = q0 + r + 8 * half;
    const int tB = tA + 16;
    const float iA = 1.0f / lA[r];
    const float iB = 1.0f / lB[r];
    dst[(size_t)tA * C + h * D +  0 + n] = (bf16)(oA0[r] * iA);
    dst[(size_t)tA * C + h * D + 16 + n] = (bf16)(oA1[r] * iA);
    dst[(size_t)tA * C + h * D + 32 + n] = (bf16)(oA2[r] * iA);
    dst[(size_t)tA * C + h * D + 48 + n] = (bf16)(oA3[r] * iA);
    dst[(size_t)tB * C + h * D +  0 + n] = (bf16)(oB0[r] * iB);
    dst[(size_t)tB * C + h * D + 16 + n] = (bf16)(oB1[r] * iB);
    dst[(size_t)tB * C + h * D + 32 + n] = (bf16)(oB2[r] * iB);
    dst[(size_t)tB * C + h * D + 48 + n] = (bf16)(oB3[r] * iB);
  }
}

// ---------------------------------------------------------------------------
// Output projection: y = AO @ Wp^T + bp, f32 output. 16x64 tile per wave,
// 2-stage register pipeline.
// ---------------------------------------------------------------------------
__global__ void __launch_bounds__(32)
out_proj(const bf16* __restrict__ AO, const bf16* __restrict__ Wp,
         const float* __restrict__ bp, float* __restrict__ out) {
  const int lane = threadIdx.x & 31;
  const int n    = lane & 15;
  const int half = lane >> 4;
  const int m0   = blockIdx.x * 16;
  const int n0   = blockIdx.y * 64;

  const bf16* arow = AO + (size_t)(m0 + n) * C;
  const bf16* b0   = Wp + (size_t)(n0 +  0 + n) * C;
  const bf16* b1   = Wp + (size_t)(n0 + 16 + n) * C;
  const bf16* b2   = Wp + (size_t)(n0 + 32 + n) * C;
  const bf16* b3   = Wp + (size_t)(n0 + 48 + n) * C;

  f32x8 acc0 = {}, acc1 = {}, acc2 = {}, acc3 = {};
  bf16x16 a  = load_a_frag(arow, lane);
  bf16x16 f0 = load_b_frag(b0, lane);
  bf16x16 f1 = load_b_frag(b1, lane);
  bf16x16 f2 = load_b_frag(b2, lane);
  bf16x16 f3 = load_b_frag(b3, lane);
#pragma unroll 4
  for (int kk = 32; kk < C; kk += 32) {
    if ((kk & 255) == 0 && kk + 512 < C) {
      __builtin_prefetch(arow + kk + 512, 0, 3);
      __builtin_prefetch(b0 + kk + 512, 0, 3);
      __builtin_prefetch(b2 + kk + 512, 0, 3);
    }
    const bf16x16 an = load_a_frag(arow + kk, lane);
    const bf16x16 g0 = load_b_frag(b0 + kk, lane);
    const bf16x16 g1 = load_b_frag(b1 + kk, lane);
    const bf16x16 g2 = load_b_frag(b2 + kk, lane);
    const bf16x16 g3 = load_b_frag(b3 + kk, lane);
    acc0 = wmma_bf16(a, f0, acc0);
    acc1 = wmma_bf16(a, f1, acc1);
    acc2 = wmma_bf16(a, f2, acc2);
    acc3 = wmma_bf16(a, f3, acc3);
    a = an; f0 = g0; f1 = g1; f2 = g2; f3 = g3;
  }
  acc0 = wmma_bf16(a, f0, acc0);
  acc1 = wmma_bf16(a, f1, acc1);
  acc2 = wmma_bf16(a, f2, acc2);
  acc3 = wmma_bf16(a, f3, acc3);

  f32x8* accs[4] = {&acc0, &acc1, &acc2, &acc3};
#pragma unroll
  for (int nb = 0; nb < 4; ++nb) {
    const int   ncol = n0 + nb * 16 + n;
    const float bv_n = bp[ncol];
#pragma unroll
    for (int r = 0; r < 8; ++r) {
      const int row = m0 + r + 8 * half;
      out[(size_t)row * C + ncol] = (*accs[nb])[r] + bv_n;
    }
  }
}

// ---------------------------------------------------------------------------
// Host launch
// ---------------------------------------------------------------------------
extern "C" void kernel_launch(void* const* d_in, const int* in_sizes, int n_in,
                              void* d_out, int out_size, void* d_ws, size_t ws_size,
                              hipStream_t stream) {
  const float* x  = (const float*)d_in[0];
  const float* Wq = (const float*)d_in[1];
  const float* bq = (const float*)d_in[2];
  const float* Wk = (const float*)d_in[3];
  const float* bk = (const float*)d_in[4];
  const float* Wv = (const float*)d_in[5];
  const float* bv = (const float*)d_in[6];
  const float* Wp = (const float*)d_in[7];
  const float* bp = (const float*)d_in[8];
  float* out = (float*)d_out;

  const size_t eX = (size_t)BT * C;  // 4,194,304 elements
  const size_t eW = (size_t)C * C;   // 1,048,576 elements

  bf16* xb  = (bf16*)d_ws;
  bf16* wqb = xb  + eX;
  bf16* wkb = wqb + eW;
  bf16* wvb = wkb + eW;
  bf16* wpb = wvb + eW;
  bf16* Qb  = wpb + eW;
  bf16* Kb  = Qb  + eX;
  bf16* Vtb = Kb  + eX;
  bf16* AOb = Vtb + eX;   // total: 5*eX + 4*eW bf16 = ~48 MB

  cvt_bf16<<<dim3(2048), dim3(256), 0, stream>>>(x,  xb,  (int)eX);
  cvt_bf16<<<dim3(1024), dim3(256), 0, stream>>>(Wq, wqb, (int)eW);
  cvt_bf16<<<dim3(1024), dim3(256), 0, stream>>>(Wk, wkb, (int)eW);
  cvt_bf16<<<dim3(1024), dim3(256), 0, stream>>>(Wv, wvb, (int)eW);
  cvt_bf16<<<dim3(1024), dim3(256), 0, stream>>>(Wp, wpb, (int)eW);

  qkv_proj<<<dim3(BT / 16, C / 64, 3), dim3(32), 0, stream>>>(
      xb, wqb, wkb, wvb, bq, bk, bv, Qb, Kb, Vtb);

  attn_fa<<<dim3(T / 32, B * H), dim3(32), 0, stream>>>(Qb, Kb, Vtb, AOb);

  out_proj<<<dim3(BT / 16, C / 64), dim3(32), 0, stream>>>(AOb, wpb, bp, out);
}
